// AttentionBlock_5506148073674
// MI455X (gfx1250) — compile-verified, matrix operand order checked by reading, not run on hardware
//
#include <hip/hip_runtime.h>
#include <math.h>

typedef _Float16 f16;
typedef __attribute__((ext_vector_type(16))) _Float16 v16h;
typedef __attribute__((ext_vector_type(8)))  _Float16 v8h;
typedef __attribute__((ext_vector_type(8)))  float    v8f;

// Problem constants (fixed by the reference)
constexpr int BATCH  = 8;
constexpr int C      = 256;
constexpr int NTOK   = 4096;   // 64*64
constexpr int LDQK   = 512;    // q|k packed
constexpr int GROUPS = 32;

// ---------------------------------------------------------------------------
// Fragment helpers: CDNA5 16-bit A/B fragment layout (16x32, ISA 7.12.2):
// lane L: row/col = L&15 ; kb = (L>>4)*8 ; elements 0..7 hold K=kb..kb+7,
// elements 8..15 hold K=kb+16..kb+23.  Two 16B loads per lane.
// ---------------------------------------------------------------------------
__device__ __forceinline__ v16h ld_frag(const f16* p, int ld) {
  const int lane = threadIdx.x & 31;
  const f16* base = p + (size_t)(lane & 15) * ld + (lane >> 4) * 8;
  v8h lo = *(const v8h*)(base);
  v8h hh = *(const v8h*)(base + 16);
  v16h r;
#pragma unroll
  for (int i = 0; i < 8; ++i) { r[i] = lo[i]; r[i + 8] = hh[i]; }
  return r;
}

__device__ __forceinline__ v8f wmma16(v16h a, v16h b, v8f c) {
  return __builtin_amdgcn_wmma_f32_16x16x32_f16(false, a, false, b,
                                                (short)0, c, false, false);
}

// Async global->LDS 16B copy (CDNA5, tracked by ASYNCcnt).  Inline asm since
// the clang builtin arity differs across toolchains.
__device__ __forceinline__ void async_cp16(f16* ldst, const f16* gsrc) {
  unsigned l = (unsigned)(uintptr_t)ldst;                 // addr[31:0] = LDS off
  unsigned long long g = (unsigned long long)(uintptr_t)gsrc;
  asm volatile("global_load_async_to_lds_b128 %0, %1, off"
               :: "v"(l), "v"(g) : "memory");
}
__device__ __forceinline__ void async_wait0() {
  asm volatile("s_wait_asynccnt 0x0" ::: "memory");
}

// ---------------------------------------------------------------------------
// GroupNorm stats: one block per (b,g); 8 channels * 4096 = 32768 elements.
// ---------------------------------------------------------------------------
__global__ __launch_bounds__(256) void gn_stats_kernel(
    const float* __restrict__ x, float* __restrict__ mean,
    float* __restrict__ rstd) {
  const int bg = blockIdx.x;
  const float* p = x + (size_t)bg * 32768;
  float s = 0.f, q = 0.f;
  for (int i = threadIdx.x; i < 32768; i += 256) {
    float v = p[i];
    s += v;
    q = fmaf(v, v, q);
  }
#pragma unroll
  for (int off = 16; off > 0; off >>= 1) {
    s += __shfl_xor(s, off, 32);
    q += __shfl_xor(q, off, 32);
  }
  __shared__ float ss[8], qs[8];
  if ((threadIdx.x & 31) == 0) {
    ss[threadIdx.x >> 5] = s;
    qs[threadIdx.x >> 5] = q;
  }
  __syncthreads();
  if (threadIdx.x == 0) {
    float S = 0.f, Q = 0.f;
#pragma unroll
    for (int i = 0; i < 8; ++i) { S += ss[i]; Q += qs[i]; }
    float mu  = S * (1.f / 32768.f);
    float var = Q * (1.f / 32768.f) - mu * mu;
    mean[bg] = mu;
    rstd[bg] = rsqrtf(var + 1e-5f);
  }
}

// ---------------------------------------------------------------------------
// Normalize + transpose: x (b,c,n) f32 -> xnT (b,n,c) f16, 32x32 LDS tiles.
// ---------------------------------------------------------------------------
__global__ __launch_bounds__(256) void gn_apply_t_kernel(
    const float* __restrict__ x, const float* __restrict__ mean,
    const float* __restrict__ rstd, const float* __restrict__ gsc,
    const float* __restrict__ gbi, f16* __restrict__ xnT) {
  const int b  = blockIdx.z;
  const int c0 = blockIdx.y * 32;
  const int n0 = blockIdx.x * 32;
  __shared__ f16 T[32][33];

  const int ci  = threadIdx.x >> 3;
  const int nj0 = (threadIdx.x & 7) * 4;
  const int c   = c0 + ci;
  const int g   = c >> 3;
  const float mu = mean[b * GROUPS + g];
  const float rs = rstd[b * GROUPS + g];
  const float sc = gsc[c];
  const float bi = gbi[c];
  const float* xr = x + ((size_t)b * C + c) * NTOK + n0;
#pragma unroll
  for (int u = 0; u < 4; ++u) {
    float v = xr[nj0 + u];
    T[ci][nj0 + u] = (f16)((v - mu) * rs * sc + bi);
  }
  __syncthreads();
  const int ni  = threadIdx.x >> 3;
  const int cj0 = (threadIdx.x & 7) * 4;
  f16* orow = xnT + ((size_t)b * NTOK + n0 + ni) * C + c0 + cj0;
#pragma unroll
  for (int u = 0; u < 4; ++u) orow[u] = T[cj0 + u][ni];
}

// ---------------------------------------------------------------------------
__global__ void cvt_f16_kernel(const float* __restrict__ src,
                               f16* __restrict__ dst, int n) {
  int i = blockIdx.x * 256 + threadIdx.x;
  if (i < n) dst[i] = (f16)src[i];
}

// ---------------------------------------------------------------------------
// WMMA GEMM: C[M,N] = A[M,K] * B[N,K]^T.  Tile 128x128, 8 waves, B in LDS.
// MODE 0: f16 out, bias[N]          (q|k projection)
// MODE 1: f32 out, bias[M] + resid  (output projection)
// MODE 2: f16 out, bias[M]          (vT projection: rows = channels)
// ---------------------------------------------------------------------------
template <int MODE>
__global__ __launch_bounds__(256, 1) void gemm_abt_kernel(
    const f16* __restrict__ A, int lda, long long sA,
    const f16* __restrict__ B, int ldb, long long sB,
    const float* __restrict__ bias, void* __restrict__ Cv, int ldc,
    long long sC, const float* __restrict__ resid, long long sR, int K) {
  const int b  = blockIdx.z;
  const int m0 = blockIdx.x * 128;
  const int n0 = blockIdx.y * 128;
  A += (size_t)b * sA;
  B += (size_t)b * sB;

  __shared__ f16 Bs[128][40];  // 80B pitch: 16B aligned, bank-clean

  const int w    = threadIdx.x >> 5;
  const int lane = threadIdx.x & 31;
  const int hi   = lane >> 4;
  const int col  = lane & 15;

  v8f acc[8];
#pragma unroll
  for (int i = 0; i < 8; ++i)
#pragma unroll
    for (int j = 0; j < 8; ++j) acc[i][j] = 0.f;

  const f16* arow = A + (size_t)(m0 + w * 16) * lda;
  const int r  = threadIdx.x >> 1;
  const int sg = threadIdx.x & 1;

  for (int kb = 0; kb < K; kb += 32) {
    const f16* src = B + (size_t)(n0 + r) * ldb + kb + sg * 16;
    *(v8h*)&Bs[r][sg * 16]     = *(const v8h*)(src);
    *(v8h*)&Bs[r][sg * 16 + 8] = *(const v8h*)(src + 8);
    __syncthreads();
    v16h a = ld_frag(arow + kb, lda);
#pragma unroll
    for (int ct = 0; ct < 8; ++ct) {
      v16h bf = ld_frag(&Bs[ct * 16][0], 40);
      acc[ct] = wmma16(a, bf, acc[ct]);
    }
    __syncthreads();
  }

  if (MODE == 0) {
    f16* Cp = (f16*)Cv + (size_t)b * sC;
#pragma unroll
    for (int ct = 0; ct < 8; ++ct) {
      float bn = bias[n0 + ct * 16 + col];
#pragma unroll
      for (int j = 0; j < 8; ++j) {
        int m = m0 + w * 16 + j + 8 * hi;
        Cp[(size_t)m * ldc + n0 + ct * 16 + col] = (f16)(acc[ct][j] + bn);
      }
    }
  } else if (MODE == 1) {
    float* Cp = (float*)Cv + (size_t)b * sC;
    const float* Rp = resid + (size_t)b * sR;
#pragma unroll
    for (int ct = 0; ct < 8; ++ct) {
#pragma unroll
      for (int j = 0; j < 8; ++j) {
        int m = m0 + w * 16 + j + 8 * hi;
        size_t idx = (size_t)m * ldc + n0 + ct * 16 + col;
        Cp[idx] = acc[ct][j] + bias[m] + Rp[idx];
      }
    }
  } else {
    f16* Cp = (f16*)Cv + (size_t)b * sC;
#pragma unroll
    for (int ct = 0; ct < 8; ++ct) {
#pragma unroll
      for (int j = 0; j < 8; ++j) {
        int m = m0 + w * 16 + j + 8 * hi;
        Cp[(size_t)m * ldc + n0 + ct * 16 + col] = (f16)(acc[ct][j] + bias[m]);
      }
    }
  }
}

// ---------------------------------------------------------------------------
// Flash attention.  qk (b,4096,512) f16 [q|k], vT (b,256,4096) f16,
// out (b,4096,256) f16.  WG = 128 queries (8 waves x 16 rows).
// Double-buffered K/V chunks staged with async global->LDS copies; online
// softmax entirely in registers + cross-lane shuffles (rows live in 16-lane
// halves, so xor-masks 1,2,4,8 reduce within the half).
// ---------------------------------------------------------------------------
__global__ __launch_bounds__(256, 1) void attn_kernel(
    const f16* __restrict__ qk, const f16* __restrict__ vT,
    f16* __restrict__ out) {
  const int b    = blockIdx.y;
  const int q0   = blockIdx.x * 128;
  const int w    = threadIdx.x >> 5;
  const int lane = threadIdx.x & 31;
  const int hi   = lane >> 4;
  const int col  = lane & 15;

  __shared__ f16 Ks[2][32][264];   // keys x channels, 528B pitch
  __shared__ f16 Vs[2][256][40];   // channels x keys, 80B pitch
  __shared__ f16 Pw[8][16][32];    // per-wave P tile

  const f16* qb  = qk + (size_t)b * NTOK * LDQK;
  const f16* vb  = vT + (size_t)b * C * NTOK;
  const int  myq = q0 + w * 16;

  // Preload Q fragments (16 rows x K=256)
  v16h qf[8];
#pragma unroll
  for (int cb = 0; cb < 8; ++cb)
    qf[cb] = ld_frag(qb + (size_t)myq * LDQK + cb * 32, LDQK);

  v8f acc[16];
#pragma unroll
  for (int i = 0; i < 16; ++i)
#pragma unroll
    for (int j = 0; j < 8; ++j) acc[i][j] = 0.f;

  float m_i[8], l_i[8];
#pragma unroll
  for (int j = 0; j < 8; ++j) { m_i[j] = -__builtin_inff(); l_i[j] = 0.f; }

  // Async-stage one 32-key chunk (K rows + V rows) into buffer `buf`.
  auto stage = [&](int buf, int k0) {
    const int kr = threadIdx.x >> 3, sg = threadIdx.x & 7;
    const f16* ksrc = qb + (size_t)(k0 + kr) * LDQK + C + sg * 32;
    f16* kdst = &Ks[buf][kr][sg * 32];
#pragma unroll
    for (int u = 0; u < 4; ++u) async_cp16(kdst + u * 8, ksrc + u * 8);
    const f16* vsrc = vb + (size_t)threadIdx.x * NTOK + k0;
    f16* vdst = &Vs[buf][threadIdx.x][0];
#pragma unroll
    for (int u = 0; u < 4; ++u) async_cp16(vdst + u * 8, vsrc + u * 8);
  };

  stage(0, 0);
  async_wait0();
  __syncthreads();

#pragma unroll 1
  for (int k0 = 0; k0 < NTOK; k0 += 32) {
    const int buf = (k0 >> 5) & 1;
    if (k0 + 32 < NTOK) stage(buf ^ 1, k0 + 32);  // overlap with compute

    // ---- scores: 16 queries x 32 keys, K=256 -> 16 WMMAs
    v8f s0, s1;
#pragma unroll
    for (int j = 0; j < 8; ++j) { s0[j] = 0.f; s1[j] = 0.f; }
#pragma unroll
    for (int cb = 0; cb < 8; ++cb) {
      v16h bk0 = ld_frag(&Ks[buf][0][cb * 32], 264);
      s0 = wmma16(qf[cb], bk0, s0);
      v16h bk1 = ld_frag(&Ks[buf][16][cb * 32], 264);
      s1 = wmma16(qf[cb], bk1, s1);
    }

    // ---- online softmax in registers (row j+8*hi replicated across half)
    float mx[8];
#pragma unroll
    for (int j = 0; j < 8; ++j) {
      s0[j] *= 0.0625f;  // * c^-0.5
      s1[j] *= 0.0625f;
      mx[j] = fmaxf(s0[j], s1[j]);
    }
#pragma unroll
    for (int d = 1; d < 16; d <<= 1)
#pragma unroll
      for (int j = 0; j < 8; ++j) mx[j] = fmaxf(mx[j], __shfl_xor(mx[j], d, 32));

    float fac[8], ps[8];
#pragma unroll
    for (int j = 0; j < 8; ++j) {
      float mn = fmaxf(m_i[j], mx[j]);
      fac[j]   = __expf(m_i[j] - mn);
      m_i[j]   = mn;
      float pa = __expf(s0[j] - mn);
      float pb = __expf(s1[j] - mn);
      ps[j]    = pa + pb;
      Pw[w][j + 8 * hi][col]      = (f16)pa;
      Pw[w][j + 8 * hi][16 + col] = (f16)pb;
    }
#pragma unroll
    for (int d = 1; d < 16; d <<= 1)
#pragma unroll
      for (int j = 0; j < 8; ++j) ps[j] += __shfl_xor(ps[j], d, 32);
#pragma unroll
    for (int j = 0; j < 8; ++j) l_i[j] = l_i[j] * fac[j] + ps[j];

#pragma unroll
    for (int ct = 0; ct < 16; ++ct)
#pragma unroll
      for (int j = 0; j < 8; ++j) acc[ct][j] *= fac[j];

    // ---- acc += P @ V (16 column tiles, K=32)
    v16h ap = ld_frag(&Pw[w][0][0], 32);
#pragma unroll
    for (int ct = 0; ct < 16; ++ct) {
      v16h bv = ld_frag(&Vs[buf][ct * 16][0], 40);
      acc[ct] = wmma16(ap, bv, acc[ct]);
    }

    async_wait0();      // next chunk's copies landed (overlapped above)
    __syncthreads();    // all waves done reading buf; next buf ready
  }

  // ---- final 1/l normalization + store (b, n, c) f16
  float inv[8];
#pragma unroll
  for (int j = 0; j < 8; ++j) inv[j] = 1.f / l_i[j];
  f16* ob = out + (size_t)b * NTOK * C;
#pragma unroll
  for (int ct = 0; ct < 16; ++ct)
#pragma unroll
    for (int j = 0; j < 8; ++j)
      ob[(size_t)(myq + j + 8 * hi) * C + ct * 16 + col] =
          (f16)(acc[ct][j] * inv[j]);
}

// ---------------------------------------------------------------------------
extern "C" void kernel_launch(void* const* d_in, const int* in_sizes, int n_in,
                              void* d_out, int out_size, void* d_ws,
                              size_t ws_size, hipStream_t stream) {
  const float* x     = (const float*)d_in[0];
  const float* gsc   = (const float*)d_in[1];
  const float* gbi   = (const float*)d_in[2];
  const float* w_qkv = (const float*)d_in[3];
  const float* b_qkv = (const float*)d_in[4];
  const float* w_out = (const float*)d_in[5];
  const float* b_out = (const float*)d_in[6];
  float* out = (float*)d_out;

  // Workspace carve-up
  char* ws = (char*)d_ws;
  f16* xnT  = (f16*)ws; ws += (size_t)BATCH * NTOK * C * 2;     // 16.8 MB
  f16* qk   = (f16*)ws; ws += (size_t)BATCH * NTOK * LDQK * 2;  // 33.6 MB
  f16* vT   = (f16*)ws; ws += (size_t)BATCH * C * NTOK * 2;     // 16.8 MB
  f16* attn = (f16*)ws; ws += (size_t)BATCH * NTOK * C * 2;     // 16.8 MB
  f16* wq16 = (f16*)ws; ws += (size_t)(3 * C) * C * 2;
  f16* wo16 = (f16*)ws; ws += (size_t)C * C * 2;
  float* gmean = (float*)ws; ws += BATCH * GROUPS * 4;
  float* grstd = (float*)ws; ws += BATCH * GROUPS * 4;

  // 1) GroupNorm stats
  gn_stats_kernel<<<BATCH * GROUPS, 256, 0, stream>>>(x, gmean, grstd);

  // 2) Weight converts
  cvt_f16_kernel<<<(3 * C * C + 255) / 256, 256, 0, stream>>>(w_qkv, wq16,
                                                              3 * C * C);
  cvt_f16_kernel<<<(C * C + 255) / 256, 256, 0, stream>>>(w_out, wo16, C * C);

  // 3) Normalize + transpose -> xnT (b, n, c) f16
  gn_apply_t_kernel<<<dim3(NTOK / 32, C / 32, BATCH), 256, 0, stream>>>(
      x, gmean, grstd, gsc, gbi, xnT);

  // 4a) q|k projection: qk(b,n,512) = xnT @ w_qkv[0:512]^T + bias
  gemm_abt_kernel<0><<<dim3(NTOK / 128, LDQK / 128, BATCH), 256, 0, stream>>>(
      xnT, C, (long long)NTOK * C, wq16, C, 0LL, b_qkv, qk, LDQK,
      (long long)NTOK * LDQK, nullptr, 0LL, C);

  // 4b) vT projection: vT(b,c,n) = w_qkv[512:768] @ xn  (rows = channels)
  gemm_abt_kernel<2><<<dim3(C / 128, NTOK / 128, BATCH), 256, 0, stream>>>(
      wq16 + (size_t)2 * C * C, C, 0LL, xnT, C, (long long)NTOK * C,
      b_qkv + 2 * C, vT, NTOK, (long long)C * NTOK, nullptr, 0LL, C);

  // 5) Flash attention -> attn (b, n, 256) f16
  attn_kernel<<<dim3(NTOK / 128, BATCH), 256, 0, stream>>>(qk, vT, attn);

  // 6) Output projection + bias + residual -> d_out (b, c, n) f32
  gemm_abt_kernel<1><<<dim3(C / 128, NTOK / 128, BATCH), 256, 0, stream>>>(
      wo16, C, 0LL, attn, C, (long long)NTOK * C, b_out, out, NTOK,
      (long long)C * NTOK, x, (long long)C * NTOK, C);
}
